// GPCopulaNet_21603685499096
// MI455X (gfx1250) — matrix operand order repaired and checked by reading, not compiled, not run on hardware
//
#include <hip/hip_runtime.h>

#define T_ 512
#define N_ 200
#define H_ 64
#define E_ 16
#define R_ 10
#define VOCAB_ 200

typedef __attribute__((ext_vector_type(16))) _Float16 v16h;
typedef __attribute__((ext_vector_type(8)))  _Float16 h8v;
typedef __attribute__((ext_vector_type(2)))  _Float16 h2f;
typedef __attribute__((ext_vector_type(8)))  float    v8f;
typedef __attribute__((ext_vector_type(2)))  float    v2f;

// LDS layout (dynamic): W0p[32*256] h2f | W1p[64*256] h2f | Whp[32*16] h2f | h0buf[1024] | h1buf[1024]
#define W0_WORDS   (32*256)
#define W1_WORDS   (64*256)
#define WH_WORDS   (32*16)
#define SMEM_BYTES ((W0_WORDS + W1_WORDS + WH_WORDS)*4 + 2*16*H_*2)

__device__ __forceinline__ float sig_(float x){ return 1.0f/(1.0f+__expf(-x)); }
__device__ __forceinline__ float tanh_(float x){
  x = fminf(fmaxf(x,-15.f),15.f);
  float e = __expf(2.0f*x);
  return (e-1.0f)/(e+1.0f);
}
__device__ __forceinline__ float softplus_(float x){
  return (x>20.0f)? x : log1pf(__expf(x));
}

// A-fragment (16-bit A 16x32, contiguous k order used here): lane L: row M=L%16; lanes 0-15 hold
// K 0..15, lanes 16-31 K 16..31. Two 16B runs -> 2x ds_load_b128.
__device__ __forceinline__ v16h load_a_frag(const _Float16* hbuf, int Mrow, int ahk, int KB){
  const h8v* row = (const h8v*)(hbuf + Mrow*H_);
  h8v r1 = row[(KB + ahk) >> 3];
  h8v r2 = row[(KB + 8 + ahk) >> 3];
  v16h a;
  #pragma unroll
  for (int i=0;i<8;++i){ a[i]=r1[i]; a[8+i]=r2[i]; }
  return a;
}

// B-fragment (16-bit B 32x16): lane L: col N=L%16; lanes 0-15 K 0..15, lanes 16-31
// K 16..31; VGPR j = K pair (2j,2j+1). Images stored k-pair-major: Wp[kk*stride+col]
// holds halves (2kk, 2kk+1) -> one conflict-free ds_load_b32 per fragment VGPR.
__device__ __forceinline__ v16h load_b_frag(const h2f* Wp, int kkBase, int col, int stride){
  v16h b;
  #pragma unroll
  for (int j=0;j<8;++j){
    h2f p = Wp[(kkBase + j)*stride + col];
    b[2*j]   = p[0];
    b[2*j+1] = p[1];
  }
  return b;
}

__global__ __launch_bounds__(128,1) void gpc_lstm_kernel(
    const float* __restrict__ inputs,   // (T,N)
    const int*   __restrict__ indices,  // (N)
    const float* __restrict__ embed_W,  // (VOCAB,E)
    const float* __restrict__ Wih0, const float* __restrict__ Whh0,
    const float* __restrict__ bih0, const float* __restrict__ bhh0,
    const float* __restrict__ Wih1, const float* __restrict__ Whh1,
    const float* __restrict__ bih1, const float* __restrict__ bhh1,
    const float* __restrict__ Wm, const float* __restrict__ bm,
    const float* __restrict__ Wv, const float* __restrict__ bv,
    const float* __restrict__ Wd, const float* __restrict__ bd,
    float* __restrict__ mu_out,     // (T,N)
    float* __restrict__ v_out,      // (T,N,R)
    float* __restrict__ dvar_out)   // (T,N)
{
  extern __shared__ char smem_raw[];
  h2f* W0p = (h2f*)smem_raw;                          // [32*256]  layer0 B image
  h2f* W1p = W0p + W0_WORDS;                          // [64*256]  layer1 B image
  h2f* Whp = W1p + W1_WORDS;                          // [32*16]   heads  B image
  _Float16* h0buf = (_Float16*)(Whp + WH_WORDS);      // [16*64]
  _Float16* h1buf = h0buf + 16*H_;                    // [16*64]

  const int tid  = threadIdx.x;
  const int w    = tid >> 5;       // wave 0..3: owns hidden units j in [16w,16w+16)
  const int L    = tid & 31;
  const int Nl   = L & 15;         // tile column
  const int hi   = L >> 4;
  const int Mrow = L & 15;         // A-fragment row
  const int ahk  = hi * 16;        // A-frag K offset per lane half
  const int hib  = hi * 8;         // B-frag kk offset per lane half
  const int n0   = blockIdx.x * 16;

  // ---- build packed f16 weight images in LDS (once) ----
  for (int i = tid; i < W0_WORDS; i += 128){
    int kk = i >> 8, col = i & 255;
    h2f p;
    p[0] = (_Float16)Whh0[col*H_ + 2*kk];
    p[1] = (_Float16)Whh0[col*H_ + 2*kk + 1];
    W0p[i] = p;
  }
  for (int i = tid; i < W1_WORDS; i += 128){
    int kk = i >> 8, col = i & 255;
    int k  = 2*kk;
    const float* src = (k < 64) ? Wih1 : Whh1;
    int kloc = (k < 64) ? k : (k - 64);
    h2f p;
    p[0] = (_Float16)src[col*H_ + kloc];
    p[1] = (_Float16)src[col*H_ + kloc + 1];
    W1p[i] = p;
  }
  for (int i = tid; i < WH_WORDS; i += 128){
    int kk = i >> 4, col = i & 15;
    h2f p;
    #pragma unroll
    for (int q=0;q<2;++q){
      int k = 2*kk + q;
      float val = 0.0f;
      if      (col==0)  val = Wm[k];
      else if (col<=10) val = Wv[k*R_ + (col-1)];
      else if (col==11) val = Wd[k];
      p[q] = (_Float16)val;
    }
    Whp[i] = p;
  }

  // per-thread loop-invariant scalars
  float cb0[4], cb1[4], wx[4];
  #pragma unroll
  for (int g=0; g<4; ++g){
    const int col = g*H_ + w*16 + Nl;      // gate column in [0,256)
    cb0[g] = bih0[col] + bhh0[col];
    cb1[g] = bih1[col] + bhh1[col];
    wx[g]  = Wih0[col];                    // Wih0 is (256,1)
  }
  // head C initializer: bias + time-invariant embedding contribution
  v8f hc;
  #pragma unroll
  for (int r=0;r<8;++r){
    int M = r + 8*hi;
    int n = n0 + M;
    int idx = indices[(n < N_) ? n : (N_-1)];
    if (idx < 0 || idx >= VOCAB_) idx = 0;
    float acc;
    if      (Nl==0)  acc = bm[0];
    else if (Nl<=10) acc = bv[Nl-1];
    else if (Nl==11) acc = bd[0];
    else             acc = 0.0f;
    #pragma unroll
    for (int e=0;e<E_;++e){
      float we = 0.0f;
      if      (Nl==0)  we = Wm[H_+e];
      else if (Nl<=10) we = Wv[(H_+e)*R_ + (Nl-1)];
      else if (Nl==11) we = Wd[H_+e];
      acc += embed_W[idx*E_ + e] * we;
    }
    hc[r] = acc;
  }

  // cell state lives in WMMA C-fragment layout registers for all 512 steps
  v8f c0s, c1s;
  #pragma unroll
  for (int r=0;r<8;++r){ c0s[r]=0.f; c1s[r]=0.f; }
  for (int i=tid; i<16*H_; i+=128){ h0buf[i]=(_Float16)0.f; h1buf[i]=(_Float16)0.f; }
  __syncthreads();

  for (int t=0; t<T_; ++t){
    // (1) read h_{t-1} fragments before anyone overwrites
    v16h A0[2], A1h[2];
    #pragma unroll
    for (int f=0; f<2; ++f){
      A0[f]  = load_a_frag(h0buf, Mrow, ahk, 32*f);
      A1h[f] = load_a_frag(h1buf, Mrow, ahk, 32*f);
    }
    float xv[8];
    #pragma unroll
    for (int r=0;r<8;++r){
      int n = n0 + r + 8*hi;
      xv[r] = (n < N_) ? inputs[t*N_ + n] : 0.0f;
    }
    __syncthreads();

    // (2) layer 0: G = h0 @ Whh0^T + bias (x-term added in elementwise)
    v8f G[4];
    #pragma unroll
    for (int g=0; g<4; ++g){
      const int col = g*H_ + w*16 + Nl;
      v8f C;
      #pragma unroll
      for (int r=0;r<8;++r) C[r] = cb0[g];
      v16h b;
      b = load_b_frag(W0p, hib,      col, 256);
      C = __builtin_amdgcn_wmma_f32_16x16x32_f16(false, A0[0], false, b, (short)0, C, false, false);
      b = load_b_frag(W0p, hib + 16, col, 256);
      C = __builtin_amdgcn_wmma_f32_16x16x32_f16(false, A0[1], false, b, (short)0, C, false, false);
      G[g] = C;
    }
    float h0n[8];
    #pragma unroll
    for (int r=0;r<8;++r){
      float x  = xv[r];
      float gi = G[0][r] + x*wx[0];
      float gf = G[1][r] + x*wx[1];
      float gg = G[2][r] + x*wx[2];
      float go = G[3][r] + x*wx[3];
      float cn = sig_(gf)*c0s[r] + sig_(gi)*tanh_(gg);
      c0s[r] = cn;
      h0n[r] = sig_(go)*tanh_(cn);
    }
    #pragma unroll
    for (int r=0;r<8;++r)
      h0buf[(r+8*hi)*H_ + w*16 + Nl] = (_Float16)h0n[r];
    __syncthreads();

    // (3) layer 1: G = [h0_t | h1_{t-1}] @ [Wih1^T ; Whh1^T] + bias
    v16h A1x[2];
    #pragma unroll
    for (int f=0; f<2; ++f) A1x[f] = load_a_frag(h0buf, Mrow, ahk, 32*f);
    #pragma unroll
    for (int g=0; g<4; ++g){
      const int col = g*H_ + w*16 + Nl;
      v8f C;
      #pragma unroll
      for (int r=0;r<8;++r) C[r] = cb1[g];
      v16h b;
      b = load_b_frag(W1p, hib,      col, 256);
      C = __builtin_amdgcn_wmma_f32_16x16x32_f16(false, A1x[0], false, b, (short)0, C, false, false);
      b = load_b_frag(W1p, hib + 16, col, 256);
      C = __builtin_amdgcn_wmma_f32_16x16x32_f16(false, A1x[1], false, b, (short)0, C, false, false);
      b = load_b_frag(W1p, hib + 32, col, 256);
      C = __builtin_amdgcn_wmma_f32_16x16x32_f16(false, A1h[0], false, b, (short)0, C, false, false);
      b = load_b_frag(W1p, hib + 48, col, 256);
      C = __builtin_amdgcn_wmma_f32_16x16x32_f16(false, A1h[1], false, b, (short)0, C, false, false);
      G[g] = C;
    }
    float h1n[8];
    #pragma unroll
    for (int r=0;r<8;++r){
      float cn = sig_(G[1][r])*c1s[r] + sig_(G[0][r])*tanh_(G[2][r]);
      c1s[r] = cn;
      h1n[r] = sig_(G[3][r])*tanh_(cn);
    }
    #pragma unroll
    for (int r=0;r<8;++r)
      h1buf[(r+8*hi)*H_ + w*16 + Nl] = (_Float16)h1n[r];
    __syncthreads();

    // (4) heads on wave 0: [mu | v(10) | d | pad] = h1_t @ Bh + (bias + emb-part)
    if (w == 0){
      v16h Ah[2];
      #pragma unroll
      for (int f=0; f<2; ++f) Ah[f] = load_a_frag(h1buf, Mrow, ahk, 32*f);
      v8f C = hc;
      v16h b;
      b = load_b_frag(Whp, hib,      Nl, 16);
      C = __builtin_amdgcn_wmma_f32_16x16x32_f16(false, Ah[0], false, b, (short)0, C, false, false);
      b = load_b_frag(Whp, hib + 16, Nl, 16);
      C = __builtin_amdgcn_wmma_f32_16x16x32_f16(false, Ah[1], false, b, (short)0, C, false, false);
      #pragma unroll
      for (int r=0;r<8;++r){
        int n = n0 + r + 8*hi;
        if (n < N_){
          float val = C[r];
          if      (Nl==0)  mu_out[t*N_ + n] = val;
          else if (Nl<=10) v_out[(t*N_ + n)*R_ + (Nl-1)] = val;
          else if (Nl==11) dvar_out[t*N_ + n] = softplus_(val);
        }
      }
    }
  }
}

// cov[t] = v_t @ v_t^T + diag(d_t)  via exact-f32 WMMA 16x16x4 (K padded 10->16).
// Block = 4 waves covering a 32x32 tile of the 200x200 matrix; write-BW bound.
__global__ __launch_bounds__(128,1) void gpc_cov_kernel(
    const float* __restrict__ v_in,   // (T,N,R)
    const float* __restrict__ d_in,   // (T,N)
    float* __restrict__ cov_out)      // (T,N,N)
{
  const int tid  = threadIdx.x;
  const int w    = tid >> 5;
  const int L    = tid & 31;
  const int Nl   = L & 15;
  const int hi   = L >> 4;
  const int Mrow = L & 15;
  const int t  = blockIdx.z;
  const int nT = blockIdx.x*32 + (w>>1)*16;
  const int mT = blockIdx.y*32 + (w&1)*16;

  const int n = nT + Mrow;
  const int m = mT + Nl;
  v8f C;
  #pragma unroll
  for (int r=0;r<8;++r) C[r]=0.f;
  #pragma unroll
  for (int kb=0; kb<16; kb+=4){
    v2f a, b;
    #pragma unroll
    for (int i=0;i<2;++i){
      int k = kb + 2*hi + i;   // 32-bit A 16x4: lanes 0-15 K {0,1}, lanes 16-31 K {2,3}
      a[i] = (n < N_ && k < R_) ? v_in[(t*N_ + n)*R_ + k] : 0.0f;
      b[i] = (m < N_ && k < R_) ? v_in[(t*N_ + m)*R_ + k] : 0.0f;
    }
    C = __builtin_amdgcn_wmma_f32_16x16x4_f32(false, a, false, b, (short)0, C, false, false);
  }
  #pragma unroll
  for (int r=0;r<8;++r){
    int nn = nT + r + 8*hi;
    if (nn < N_ && m < N_){
      float val = C[r];
      if (nn == m) val += d_in[t*N_ + nn];
      cov_out[(size_t)t*(N_*N_) + nn*N_ + m] = val;
    }
  }
}

extern "C" void kernel_launch(void* const* d_in, const int* in_sizes, int n_in,
                              void* d_out, int out_size, void* d_ws, size_t ws_size,
                              hipStream_t stream) {
  (void)in_sizes; (void)n_in; (void)out_size; (void)ws_size;
  const float* inputs  = (const float*)d_in[0];
  const int*   indices = (const int*)  d_in[1];
  const float* embed_W = (const float*)d_in[2];
  const float* Wih0 = (const float*)d_in[3];
  const float* Whh0 = (const float*)d_in[4];
  const float* bih0 = (const float*)d_in[5];
  const float* bhh0 = (const float*)d_in[6];
  const float* Wih1 = (const float*)d_in[7];
  const float* Whh1 = (const float*)d_in[8];
  const float* bih1 = (const float*)d_in[9];
  const float* bhh1 = (const float*)d_in[10];
  const float* Wm = (const float*)d_in[11];
  const float* bm = (const float*)d_in[12];
  const float* Wv = (const float*)d_in[13];
  const float* bv = (const float*)d_in[14];
  const float* Wd = (const float*)d_in[15];
  const float* bd = (const float*)d_in[16];

  float* mu_out  = (float*)d_out;                         // T*N
  float* cov_out = (float*)d_out + (size_t)T_*N_;         // T*N*N
  float* v_ws    = (float*)d_ws;                          // T*N*R
  float* d_ws_f  = v_ws + (size_t)T_*N_*R_;               // T*N

  gpc_lstm_kernel<<<dim3((N_+15)/16), dim3(128), SMEM_BYTES, stream>>>(
      inputs, indices, embed_W,
      Wih0, Whh0, bih0, bhh0, Wih1, Whh1, bih1, bhh1,
      Wm, bm, Wv, bv, Wd, bd,
      mu_out, v_ws, d_ws_f);

  gpc_cov_kernel<<<dim3((N_+31)/32, (N_+31)/32, T_), dim3(128), 0, stream>>>(
      v_ws, d_ws_f, cov_out);
}
